// AutocompleteModel_28166395527681
// MI455X (gfx1250) — matrix operand order repaired and checked by reading, not compiled, not run on hardware
//
#include <hip/hip_runtime.h>
#include <math.h>

// ---------------------------------------------------------------------------
// AutocompleteModel on MI455X (gfx1250), wave32, WMMA bf16 + TDM.
//   x[64,512,768] -> 2-layer LSTM(H=256) -> LayerNorm -> attn-pool -> FC(30522)
// GEMMs use v_wmma_f32_16x16x32_bf16. LSTM recurrence is a persistent
// 16-workgroup kernel: W_hh sharded into LDS via tensor_load_to_lds, the
// hidden-state tile DMA'd into LDS each step via TDM, L2-atomic grid barriers.
// ---------------------------------------------------------------------------

typedef __bf16 bf16;
typedef bf16  v16bf __attribute__((ext_vector_type(16)));
typedef bf16  v8bf  __attribute__((ext_vector_type(8)));
typedef bf16  v4bf  __attribute__((ext_vector_type(4)));
typedef float v8f   __attribute__((ext_vector_type(8)));
typedef unsigned int u32x4 __attribute__((ext_vector_type(4)));
typedef int   i32x8 __attribute__((ext_vector_type(8)));
typedef int   i32x4 __attribute__((ext_vector_type(4)));

#define WMMA_BF16(a, b, c) \
  __builtin_amdgcn_wmma_f32_16x16x32_bf16(false, (a), false, (b), (short)0, (c), false, false)

// A fragment: lane (m = lane&15, lh = lane>>4) holds K = {koff..koff+7, koff+16..koff+23},
// koff = 8*lh.  p must point at (row, kk + koff).
__device__ __forceinline__ v16bf make_afrag(const bf16* p) {
  v8bf lo = *(const v8bf*)(p);
  v8bf hi = *(const v8bf*)(p + 16);
  return __builtin_shufflevector(lo, hi, 0,1,2,3,4,5,6,7,8,9,10,11,12,13,14,15);
}
// B fragment: lane n holds K = {koff..koff+7, koff+16..koff+23} of one row of row-major W.
__device__ __forceinline__ v16bf load16(const bf16* p) {
  v8bf lo = *(const v8bf*)(p);
  v8bf hi = *(const v8bf*)(p + 16);
  return __builtin_shufflevector(lo, hi, 0,1,2,3,4,5,6,7,8,9,10,11,12,13,14,15);
}

__device__ __forceinline__ float sigmoidf_(float x) { return 1.0f / (1.0f + __expf(-x)); }

// ---------------------------------------------------------------------------
// Tensor Data Mover: DMA a 2-D bf16 tile [rows x cols] (row-major, stride in
// elements) from global memory into LDS. D# layout per CDNA5 ISA ch.8.
// Issue from one wave; caller must s_wait_tensorcnt + barrier before use.
// 6-arg builtin (clang-23 / therock toolchain): extra i32x8 group, zero-filled.
// ---------------------------------------------------------------------------
__device__ __forceinline__ void tdm_load_2d_bf16(const bf16* gptr, bf16* lptr,
                                                 int rows, int cols, int stride) {
  const unsigned long long ga = (unsigned long long)(size_t)gptr;
  const unsigned lds = (unsigned)(size_t)lptr;  // flat->LDS: low 32 bits
  u32x4 g0;
  g0[0] = 1u;                                        // count=1, user descriptor
  g0[1] = lds;                                       // lds_addr
  g0[2] = (unsigned)(ga & 0xFFFFFFFFull);            // global_addr[31:0]
  g0[3] = (unsigned)((ga >> 32) & 0x1FFFFFFull)      // global_addr[56:32]
          | (2u << 30);                              // type = 2 ("image")
  i32x8 g1;
  g1[0] = 0x00010000;                                // data_size=2B, mask=0
  g1[1] = (cols & 0xFFFF) << 16;                     // tensor_dim0[15:0]
  g1[2] = ((unsigned)cols >> 16) | ((rows & 0xFFFF) << 16);  // dim0 hi | dim1 lo
  g1[3] = ((unsigned)rows >> 16) | ((cols & 0xFFFF) << 16);  // dim1 hi | tile_dim0
  g1[4] = rows & 0xFFFF;                             // tile_dim1 (tile_dim2=0)
  g1[5] = stride;                                    // tensor_dim0_stride[31:0]
  g1[6] = 0;
  g1[7] = 0;
  i32x4 g2; g2[0] = 0; g2[1] = 0; g2[2] = 0; g2[3] = 0;
  i32x4 g3; g3[0] = 0; g3[1] = 0; g3[2] = 0; g3[3] = 0;
  i32x8 g4; g4[0] = 0; g4[1] = 0; g4[2] = 0; g4[3] = 0;
  g4[4] = 0; g4[5] = 0; g4[6] = 0; g4[7] = 0;
  __builtin_amdgcn_tensor_load_to_lds(g0, g1, g2, g3, g4, 0);
}

// ---------------------------------------------------------------------------
// Generic GEMM:  C[M,N] = A[M,K](bf16,row-major) * W[N,K](bf16,row-major)^T + bias
// Wave tile 64x32 (4 M-tiles x 2 N-tiles), WG = 8 waves arranged WM x WN.
// ---------------------------------------------------------------------------
template <int WM, int WN>
__global__ __launch_bounds__(256) void gemm_bf16_nt(
    const bf16* __restrict__ A, const bf16* __restrict__ W,
    const float* __restrict__ bias, float* __restrict__ C,
    int M, int N, int K) {
  const int lane = threadIdx.x & 31;
  const int wave = threadIdx.x >> 5;
  const int wm = wave % WM, wn = wave / WM;
  const int m0 = blockIdx.y * (64 * WM) + wm * 64;
  const int n0 = blockIdx.x * (32 * WN) + wn * 32;
  const int lm = lane & 15, lh = lane >> 4;
  const int koff = lh * 8;

  v8f acc[4][2];
#pragma unroll
  for (int i = 0; i < 4; i++)
#pragma unroll
    for (int j = 0; j < 2; j++) acc[i][j] = 0.0f;

  for (int kk = 0; kk < K; kk += 32) {
    v16bf a[4], b[2];
#pragma unroll
    for (int mt = 0; mt < 4; mt++) {
      const bf16* pa = A + (size_t)(m0 + mt * 16 + lm) * K + kk + koff;
      __builtin_prefetch(pa + 32, 0, 1);  // global_prefetch_b8 (next K tile)
      a[mt] = make_afrag(pa);
    }
#pragma unroll
    for (int nt = 0; nt < 2; nt++) {
      int bn = n0 + nt * 16 + lm;
      if (bn >= N) bn = N - 1;  // clamp for ragged N (stores are guarded)
      b[nt] = load16(W + (size_t)bn * K + kk + lh * 8);
    }
#pragma unroll
    for (int mt = 0; mt < 4; mt++)
#pragma unroll
      for (int nt = 0; nt < 2; nt++)
        acc[mt][nt] = WMMA_BF16(a[mt], b[nt], acc[mt][nt]);
  }

#pragma unroll
  for (int mt = 0; mt < 4; mt++) {
    const int rbase = m0 + mt * 16 + 8 * lh;
#pragma unroll
    for (int nt = 0; nt < 2; nt++) {
      const int col = n0 + nt * 16 + lm;
      if (col < N) {
        const float badd = bias ? bias[col] : 0.0f;
#pragma unroll
        for (int j = 0; j < 8; j++)
          C[(size_t)(rbase + j) * N + col] = acc[mt][nt][j] + badd;
      }
    }
  }
}

// ---------------------------------------------------------------------------
// Persistent LSTM recurrence. 16 WGs x 256 threads. WG r owns gate-unit rows
// [64r, 64r+64) of W_hh (LDS slab via TDM). Per step:
//   TDM: hcur [64,256] bf16 global -> LDS
//   phase 1: gate[b,u] = h(t-1) . W_hh[u,:]  (WMMA from LDS) + xg[b,t,u]
//   phase 2: c,h update (each thread owns 4 fixed (b,h) cells; c in regs)
// Grid-wide sense-reversing barrier (L2 atomics) between phases.
// ---------------------------------------------------------------------------
#define NBLK 16

__device__ __forceinline__ void grid_barrier(unsigned* cnt, unsigned* phase,
                                             unsigned* s_phase) {
  __syncthreads();
  if (threadIdx.x == 0) {
    __threadfence();
    const unsigned tgt = *s_phase + 1;
    unsigned ticket =
        __hip_atomic_fetch_add(cnt, 1u, __ATOMIC_ACQ_REL, __HIP_MEMORY_SCOPE_AGENT);
    if (ticket == NBLK - 1) {
      __hip_atomic_store(cnt, 0u, __ATOMIC_RELAXED, __HIP_MEMORY_SCOPE_AGENT);
      __hip_atomic_fetch_add(phase, 1u, __ATOMIC_RELEASE, __HIP_MEMORY_SCOPE_AGENT);
    } else {
      while (__hip_atomic_load(phase, __ATOMIC_ACQUIRE, __HIP_MEMORY_SCOPE_AGENT) < tgt)
        __builtin_amdgcn_s_sleep(1);
    }
    *s_phase = tgt;
  }
  __syncthreads();
}

template <bool OUT_BF16, bool OUT_F32>
__global__ __launch_bounds__(256) void lstm_rec(
    const float* __restrict__ xg,   // [64,512,1024]  (x*W_ih^T + b_ih + b_hh)
    const bf16* __restrict__ whh,   // [1024,256]
    float* __restrict__ gate,       // [64,1024] scratch
    bf16* __restrict__ hcur,        // [64,256]
    bf16* __restrict__ hout16,      // [64,512,256] or null
    float* __restrict__ hout32,     // [64,512,256] or null
    unsigned* __restrict__ bar_cnt, unsigned* __restrict__ bar_phase) {
  __shared__ bf16 s_w[64 * 256];    // this WG's W_hh slab
  __shared__ bf16 s_h[64 * 256];    // hidden-state tile for current step
  __shared__ unsigned s_phase;
  const int tid = threadIdx.x;
  const int wg = blockIdx.x;
  const int lane = tid & 31, wave = tid >> 5;

  // TDM: DMA this WG's W_hh slab into LDS (once).
  if (wave == 0) {
    tdm_load_2d_bf16(whh + (size_t)wg * 64 * 256, s_w, 64, 256, 256);
  }
  if (tid == 0)
    s_phase = __hip_atomic_load(bar_phase, __ATOMIC_RELAXED, __HIP_MEMORY_SCOPE_AGENT);

  // Each thread owns 4 consecutive cells of the flattened [64,256] state.
  const int e0 = (wg * 256 + tid) * 4;
  const int eb = e0 >> 8;     // batch
  const int eh = e0 & 255;    // hidden base
  float c[4] = {0.f, 0.f, 0.f, 0.f};
  {
    v4bf z;
    z[0] = (bf16)0.f; z[1] = (bf16)0.f; z[2] = (bf16)0.f; z[3] = (bf16)0.f;
    *(v4bf*)(hcur + e0) = z;  // h0 = 0
  }
  if (wave == 0) __builtin_amdgcn_s_wait_tensorcnt(0);
  grid_barrier(bar_cnt, bar_phase, &s_phase);  // hcur + LDS slab visible

  const int m0 = (wave >> 1) * 16;  // batch tile: 0,16,32,48
  const int n0 = (wave & 1) * 32;   // unit tile within slab: 0,32
  const int lm = lane & 15, lh = lane >> 4;

  for (int t = 0; t < 512; ++t) {
    // ----- TDM: pull h(t-1) [64,256] bf16 into LDS -----
    if (wave == 0) {
      tdm_load_2d_bf16(hcur, s_h, 64, 256, 256);
      __builtin_amdgcn_s_wait_tensorcnt(0);
    }
    __syncthreads();

    // ----- phase 1: gate pre-activations via WMMA (all operands in LDS) -----
    v8f acc0 = 0.0f, acc1 = 0.0f;
#pragma unroll
    for (int kk = 0; kk < 256; kk += 32) {
      v16bf a  = make_afrag(s_h + (m0 + lm) * 256 + kk + lh * 8);
      v16bf b0 = load16(s_w + (n0 + lm) * 256 + kk + lh * 8);
      v16bf b1 = load16(s_w + (n0 + 16 + lm) * 256 + kk + lh * 8);
      acc0 = WMMA_BF16(a, b0, acc0);
      acc1 = WMMA_BF16(a, b1, acc1);
    }
    {
      const int rbase = m0 + 8 * lh;
      const int u0 = wg * 64 + n0 + lm;
      const int u1 = u0 + 16;
#pragma unroll
      for (int j = 0; j < 8; j++) {
        const int b_ = rbase + j;
        const float* xgp = xg + ((size_t)b_ * 512 + t) * 1024;
        gate[(size_t)b_ * 1024 + u0] = acc0[j] + xgp[u0];
        gate[(size_t)b_ * 1024 + u1] = acc1[j] + xgp[u1];
      }
    }
    grid_barrier(bar_cnt, bar_phase, &s_phase);

    // ----- phase 2: elementwise LSTM cell update -----
    {
      const float* gp = gate + (size_t)eb * 1024;
      float hv[4];
#pragma unroll
      for (int j = 0; j < 4; j++) {
        const int hh = eh + j;
        const float ig = sigmoidf_(gp[hh]);
        const float fg = sigmoidf_(gp[256 + hh]);
        const float gg = tanhf(gp[512 + hh]);
        const float og = sigmoidf_(gp[768 + hh]);
        c[j] = fg * c[j] + ig * gg;
        hv[j] = og * tanhf(c[j]);
      }
      v4bf hb;
#pragma unroll
      for (int j = 0; j < 4; j++) hb[j] = (bf16)hv[j];
      *(v4bf*)(hcur + e0) = hb;
      if (OUT_BF16) *(v4bf*)(hout16 + ((size_t)eb * 512 + t) * 256 + eh) = hb;
      if (OUT_F32) {
#pragma unroll
        for (int j = 0; j < 4; j++)
          hout32[((size_t)eb * 512 + t) * 256 + eh + j] = hv[j];
      }
    }
    grid_barrier(bar_cnt, bar_phase, &s_phase);
  }
}

// ---------------------------------------------------------------------------
// LayerNorm (in place) + attention score per (b,t) row. Block = 256 = H.
// ---------------------------------------------------------------------------
__device__ __forceinline__ float block_reduce_sum(float v, float* red) {
  const int tid = threadIdx.x;
  red[tid] = v;
  __syncthreads();
  for (int s = 128; s > 0; s >>= 1) {
    if (tid < s) red[tid] += red[tid + s];
    __syncthreads();
  }
  const float r = red[0];
  __syncthreads();
  return r;
}

__global__ __launch_bounds__(256) void ln_score_kernel(
    float* __restrict__ h, const float* __restrict__ lnw,
    const float* __restrict__ lnb, const float* __restrict__ attnw,
    float* __restrict__ scores) {
  __shared__ float red[256];
  const int row = blockIdx.x, tid = threadIdx.x;
  const float v = h[(size_t)row * 256 + tid];
  const float mean = block_reduce_sum(v, red) * (1.0f / 256.0f);
  const float d = v - mean;
  const float var = block_reduce_sum(d * d, red) * (1.0f / 256.0f);
  const float nv = d * rsqrtf(var + 1e-5f) * lnw[tid] + lnb[tid];
  h[(size_t)row * 256 + tid] = nv;
  const float sc = block_reduce_sum(nv * attnw[tid], red);
  if (tid == 0) scores[row] = sc;
}

// ---------------------------------------------------------------------------
// Softmax over T + weighted pooling. Block per batch, thread per hidden unit.
// ---------------------------------------------------------------------------
__global__ __launch_bounds__(256) void attend_kernel(
    const float* __restrict__ normed, const float* __restrict__ scores,
    bf16* __restrict__ attended16) {
  __shared__ float w[512];
  __shared__ float red[256];
  const int b = blockIdx.x, tid = threadIdx.x;
  const float s0 = scores[b * 512 + tid];
  const float s1 = scores[b * 512 + 256 + tid];
  red[tid] = fmaxf(s0, s1);
  __syncthreads();
  for (int s = 128; s > 0; s >>= 1) {
    if (tid < s) red[tid] = fmaxf(red[tid], red[tid + s]);
    __syncthreads();
  }
  const float mx = red[0];
  __syncthreads();
  const float e0 = __expf(s0 - mx), e1 = __expf(s1 - mx);
  const float inv = 1.0f / block_reduce_sum(e0 + e1, red);
  w[tid] = e0 * inv;
  w[tid + 256] = e1 * inv;
  __syncthreads();
  float acc = 0.0f;
  for (int t = 0; t < 512; t++)
    acc += w[t] * normed[((size_t)b * 512 + t) * 256 + tid];
  attended16[b * 256 + tid] = (bf16)acc;
}

// ---------------------------------------------------------------------------
// Utility kernels.
// ---------------------------------------------------------------------------
__global__ void cvt_f32_bf16(const float* __restrict__ in, bf16* __restrict__ out, int n) {
  int i = blockIdx.x * blockDim.x + threadIdx.x;
  const int stride = gridDim.x * blockDim.x;
  for (; i < n; i += stride) out[i] = (bf16)in[i];
}

__global__ void add_vec(const float* __restrict__ a, const float* __restrict__ b,
                        float* __restrict__ out, int n) {
  const int i = blockIdx.x * blockDim.x + threadIdx.x;
  if (i < n) out[i] = a[i] + b[i];
}

// ---------------------------------------------------------------------------
extern "C" void kernel_launch(void* const* d_in, const int* in_sizes, int n_in,
                              void* d_out, int out_size, void* d_ws, size_t ws_size,
                              hipStream_t stream) {
  (void)in_sizes; (void)n_in; (void)out_size; (void)ws_size;
  const float* x     = (const float*)d_in[0];
  const float* w_ih0 = (const float*)d_in[1];
  const float* w_hh0 = (const float*)d_in[2];
  const float* b_ih0 = (const float*)d_in[3];
  const float* b_hh0 = (const float*)d_in[4];
  const float* w_ih1 = (const float*)d_in[5];
  const float* w_hh1 = (const float*)d_in[6];
  const float* b_ih1 = (const float*)d_in[7];
  const float* b_hh1 = (const float*)d_in[8];
  const float* ln_w  = (const float*)d_in[9];
  const float* ln_b  = (const float*)d_in[10];
  const float* attnw = (const float*)d_in[11];
  const float* fc_w  = (const float*)d_in[12];
  const float* fc_b  = (const float*)d_in[13];
  float* out = (float*)d_out;

  // workspace layout (deterministic cursor, 256B aligned)
  size_t off = 0;
  auto take = [&](size_t bytes) -> void* {
    void* p = (char*)d_ws + off;
    off += (bytes + 255) & ~(size_t)255;
    return p;
  };
  unsigned* bar    = (unsigned*)take(256);        // [0]=count, [32]=phase
  float* gate      = (float*)take((size_t)64 * 1024 * 4);
  bf16*  hcur      = (bf16*) take((size_t)64 * 256 * 2);
  bf16*  xb        = (bf16*) take((size_t)64 * 512 * 768 * 2);
  bf16*  wih0b     = (bf16*) take((size_t)1024 * 768 * 2);
  bf16*  whh0b     = (bf16*) take((size_t)1024 * 256 * 2);
  bf16*  wih1b     = (bf16*) take((size_t)1024 * 256 * 2);
  bf16*  whh1b     = (bf16*) take((size_t)1024 * 256 * 2);
  bf16*  fcwb      = (bf16*) take((size_t)30522 * 256 * 2);
  float* bias0     = (float*)take(1024 * 4);
  float* bias1     = (float*)take(1024 * 4);
  float* xgbuf     = (float*)take((size_t)64 * 512 * 1024 * 4);
  bf16*  h1b       = (bf16*) take((size_t)64 * 512 * 256 * 2);
  float* h2        = (float*)take((size_t)64 * 512 * 256 * 4);
  float* scores    = (float*)take((size_t)64 * 512 * 4);
  bf16*  att16     = (bf16*) take((size_t)64 * 256 * 2);

  (void)hipMemsetAsync(bar, 0, 256, stream);

  // fp32 -> bf16 staging
  cvt_f32_bf16<<<2048, 256, 0, stream>>>(x, xb, 64 * 512 * 768);
  cvt_f32_bf16<<<512, 256, 0, stream>>>(w_ih0, wih0b, 1024 * 768);
  cvt_f32_bf16<<<256, 256, 0, stream>>>(w_hh0, whh0b, 1024 * 256);
  cvt_f32_bf16<<<256, 256, 0, stream>>>(w_ih1, wih1b, 1024 * 256);
  cvt_f32_bf16<<<256, 256, 0, stream>>>(w_hh1, whh1b, 1024 * 256);
  cvt_f32_bf16<<<2048, 256, 0, stream>>>(fc_w, fcwb, 30522 * 256);
  add_vec<<<4, 256, 0, stream>>>(b_ih0, b_hh0, bias0, 1024);
  add_vec<<<4, 256, 0, stream>>>(b_ih1, b_hh1, bias1, 1024);

  // layer 0: xg = x @ w_ih0^T + (b_ih0+b_hh0)   [32768 x 1024], K=768
  gemm_bf16_nt<2, 4><<<dim3(8, 256), 256, 0, stream>>>(
      xb, wih0b, bias0, xgbuf, 32768, 1024, 768);
  lstm_rec<true, false><<<NBLK, 256, 0, stream>>>(
      xgbuf, whh0b, gate, hcur, h1b, nullptr, bar, bar + 32);

  // layer 1: xg = h1 @ w_ih1^T + (b_ih1+b_hh1)  [32768 x 1024], K=256
  gemm_bf16_nt<2, 4><<<dim3(8, 256), 256, 0, stream>>>(
      h1b, wih1b, bias1, xgbuf, 32768, 1024, 256);
  lstm_rec<false, true><<<NBLK, 256, 0, stream>>>(
      xgbuf, whh1b, gate, hcur, nullptr, h2, bar, bar + 32);

  // LayerNorm + attention pooling
  ln_score_kernel<<<32768, 256, 0, stream>>>(h2, ln_w, ln_b, attnw, scores);
  attend_kernel<<<64, 256, 0, stream>>>(h2, scores, att16);

  // final FC: [64 x 30522] = attended @ fc_w^T + fc_b, K=256
  gemm_bf16_nt<1, 8><<<dim3(120, 1), 256, 0, stream>>>(
      att16, fcwb, fc_b, out, 64, 30522, 256);
}
